// GraphAdjacencyLayer_65481071399730
// MI455X (gfx1250) — compile-verified, matrix operand order checked by reading, not run on hardware
//
#include <hip/hip_runtime.h>
#include <hip/hip_bf16.h>

// ---------------------------------------------------------------------------
// GraphAdjacencyLayer for MI455X (gfx1250, wave32, WMMA)
//
// Roofline: 2*8192^2*128 = 17.2 GFLOP vs 256 MB output store.
// Store floor @23.3 TB/s ~ 11 us  =>  need ~1.6 PFLOPS matmul to hide compute.
// => use V_WMMA_F32_16X16X32_F16 (f16 in, f32 accumulate). Normalized inputs
// are unit vectors so f16 A/B with f32 accumulation is numerically safe for
// thresholding at 0.5 / 0.9.
// ---------------------------------------------------------------------------

typedef __attribute__((ext_vector_type(16))) _Float16 v16h;
typedef __attribute__((ext_vector_type(8)))  _Float16 v8h;
typedef __attribute__((ext_vector_type(4)))  _Float16 v4h;
typedef __attribute__((ext_vector_type(8)))  float    v8f;
typedef __attribute__((ext_vector_type(4)))  float    v4f;

#define NROWS 8192
#define DDIM  128
#define THRESHOLD 0.9f
#define SECONDARY 0.5f
#define SEC_W     0.5f
#define EPS       1e-12f

// ---------------------------------------------------------------------------
// Kernel 1: row normalization fp32 -> fp16.
// One wave32 per row: 32 lanes x float4 = 128 elements. Wave shfl reduction.
// Output: fn[8192][128] as _Float16 in d_ws (2 MB, L2-resident for the GEMM).
// ---------------------------------------------------------------------------
__global__ __launch_bounds__(256) void normalize_rows_kernel(
    const float* __restrict__ feat, _Float16* __restrict__ fn) {
  const int row  = (int)((blockIdx.x * blockDim.x + threadIdx.x) >> 5);
  const int lane = (int)(threadIdx.x & 31u);
  if (row >= NROWS) return;

  const float* rp = feat + (size_t)row * DDIM + lane * 4;
  v4f v = *(const v4f*)rp;
  float ss = v.x * v.x + v.y * v.y + v.z * v.z + v.w * v.w;

  // wave32 butterfly reduction
  #pragma unroll
  for (int m = 16; m >= 1; m >>= 1) ss += __shfl_xor(ss, m, 32);

  const float scale = 1.0f / (sqrtf(ss) + EPS);

  v4h h;
  h.x = (_Float16)(v.x * scale);
  h.y = (_Float16)(v.y * scale);
  h.z = (_Float16)(v.z * scale);
  h.w = (_Float16)(v.w * scale);
  *(v4h*)(fn + (size_t)row * DDIM + lane * 4) = h;
}

// ---------------------------------------------------------------------------
// Kernel 2: C = fn * fn^T with WMMA, fused fid/threshold/diagonal epilogue.
//
// One wave per 16x16 output tile; K=128 as 4 chained v_wmma_f32_16x16x32_f16.
// A-fragment ISA layout (16-bit A, 16x32): lanes 0-15 hold row M=lane with
// K = {0..7, 16..23}; lanes 16-31 hold row M=lane-16 with K = {8..15, 24..31}.
// Since B = fn^T, the B fragment (32x16, K-major) uses the *same* per-lane
// load pattern taken from the N-block's rows of fn.
//
// C/D layout: VGPR r, lanes 0-15 -> (M=r,     N=lane),
//                     lanes 16-31 -> (M=r+8,  N=lane-16).
// ---------------------------------------------------------------------------
__global__ __launch_bounds__(256) void gram_adjacency_kernel(
    const _Float16* __restrict__ fn, float* __restrict__ out) {
  const int lane  = (int)(threadIdx.x & 31u);
  const int wave  = (int)(threadIdx.x >> 5);
  const int tileN = (int)blockIdx.x * 8 + wave;  // 0..511
  const int tileM = (int)blockIdx.y;             // 0..511

  const int  rsel = lane & 15;                 // row-within-tile this lane feeds
  const int  koff = (lane < 16) ? 0 : 8;       // half-K offset per ISA layout

  const _Float16* pa = fn + (size_t)(tileM * 16 + rsel) * DDIM + koff;
  const _Float16* pb = fn + (size_t)(tileN * 16 + rsel) * DDIM + koff;

  v8f acc = {};
  #pragma unroll
  for (int kc = 0; kc < 4; ++kc) {
    const int kb = kc * 32;
    // two 16B loads per fragment give the exact {0..7,16..23}/{8..15,24..31}
    // per-lane K packing
    v8h a0 = *(const v8h*)(pa + kb);
    v8h a1 = *(const v8h*)(pa + kb + 16);
    v8h b0 = *(const v8h*)(pb + kb);
    v8h b1 = *(const v8h*)(pb + kb + 16);
    v16h a = __builtin_shufflevector(a0, a1, 0, 1, 2, 3, 4, 5, 6, 7,
                                             8, 9, 10, 11, 12, 13, 14, 15);
    v16h b = __builtin_shufflevector(b0, b1, 0, 1, 2, 3, 4, 5, 6, 7,
                                             8, 9, 10, 11, 12, 13, 14, 15);
    // (neg_a, A, neg_b, B, c_mod, C, reuse_a, reuse_b)
    acc = __builtin_amdgcn_wmma_f32_16x16x32_f16(
        false, a, false, b, (short)0, acc, false, false);
  }

  // Fused epilogue: fid = gram^2, two-level threshold, zero diagonal.
  // Output is 256 MB streaming (> 192 MB L2) -> nontemporal stores.
  const int col = tileN * 16 + (lane & 15);
  const int mhi = (lane < 16) ? 0 : 8;
  #pragma unroll
  for (int r = 0; r < 8; ++r) {
    const int row = tileM * 16 + mhi + r;
    const float g   = acc[r];
    const float fid = g * g;
    float w = (fid >= THRESHOLD) ? 1.0f : ((fid >= SECONDARY) ? SEC_W : 0.0f);
    if (row == col) w = 0.0f;
    __builtin_nontemporal_store(w, &out[(size_t)row * NROWS + col]);
  }
}

extern "C" void kernel_launch(void* const* d_in, const int* in_sizes, int n_in,
                              void* d_out, int out_size, void* d_ws, size_t ws_size,
                              hipStream_t stream) {
  (void)in_sizes; (void)n_in; (void)out_size; (void)ws_size;

  const float* feat = (const float*)d_in[0];
  float*       out  = (float*)d_out;
  _Float16*    fn   = (_Float16*)d_ws;  // 8192*128*2 B = 2 MB scratch

  // 8192 rows, 8 waves (rows) per 256-thread block -> 1024 blocks
  normalize_rows_kernel<<<dim3(NROWS / 8), 256, 0, stream>>>(feat, fn);

  // 512x512 tiles of 16x16; 8 waves per block = 8 N-tiles per block
  gram_adjacency_kernel<<<dim3(512 / 8, 512), 256, 0, stream>>>(fn, out);
}
